// ComplexAttention_88819923681503
// MI455X (gfx1250) — matrix operand order repaired and checked by reading, not compiled, not run on hardware
//
#include <hip/hip_runtime.h>
#include <hip/hip_bf16.h>

typedef _Float16 half_t;
typedef __attribute__((ext_vector_type(16))) _Float16 v16h;
typedef __attribute__((ext_vector_type(8)))  _Float16 v8h;
typedef __attribute__((ext_vector_type(8)))  float    v8f;
typedef __attribute__((ext_vector_type(4)))  float    v4f;

#define DIMN   1024
#define HEADS  16
#define HDIM   64
#define BATCH  4
#define SEQ    2048
#define MTOT   (BATCH*SEQ)          // 8192
#define SCALEQ 0.125f               // HD^-0.5

union Frag16 { v16h v; v8h h[2]; };

__device__ inline v8f wmma_f16(v16h a, v16h b, v8f c) {
    return __builtin_amdgcn_wmma_f32_16x16x32_f16(false, a, false, b, (short)0, c,
                                                  false, false);
}

// ---------------------------------------------------------------------------
// Generic projection GEMM: C[M=8192, N=1024] = X[M,K=1024] * W[N,K]^T + bias
// Double-buffered LDS: staging of tile k+1 overlaps WMMAs of tile k.
// MODE 0: out f16 [B,H,L,HD]   (Q/K, optionally pre-scaled)
// MODE 1: out f16 [B,H,HD,L]   (V transposed for attention B-fragments)
// MODE 2: out f32 [M,N]        (final output projection)
// XHALF : A operand is f16 -> staged via GLOBAL_LOAD_ASYNC_TO_LDS_B128
// ---------------------------------------------------------------------------
struct GemmJob {
    const void*  x;       // f32 or f16 [M,K]
    const float* W;       // f32 [N,K]
    const float* bias;    // f32 [N]
    void*        out;
    float        scale;
};
struct GemmArgs { GemmJob job[6]; };

template <int MODE, bool XHALF>
__global__ __launch_bounds__(256)
void proj_gemm(GemmArgs args) {
    const GemmJob jb = args.job[blockIdx.z];

    __shared__ half_t As[2][128 * 40];   // 128 rows x 32 halves, padded to 40
    __shared__ half_t Bs[2][128 * 40];

    const int tid  = threadIdx.x;
    const int lane = tid & 31;
    const int wave = tid >> 5;
    const int ll   = lane & 15;
    const int lh   = lane >> 4;
    const int wm   = (wave & 3) * 32;   // wave M offset within block tile
    const int wn   = (wave >> 2) * 64;  // wave N offset within block tile
    const int mBlock = blockIdx.y * 128;
    const int nBlock = blockIdx.x * 128;

    // cooperative load mapping: 2 threads per row, 16 elements each
    const int grow = tid >> 1;
    const int gcol = (tid & 1) * 16;
    const unsigned ldsA[2] = {                       // LDS byte offsets (async dst)
        (unsigned)(size_t)&As[0][grow * 40 + gcol],
        (unsigned)(size_t)&As[1][grow * 40 + gcol]
    };

    auto stage = [&](int buf, int k0) {
        // ---- X tile ----
        if (XHALF) {
            // f16 activations: DMA straight to LDS, no VGPR round-trip
            const half_t* xp = (const half_t*)jb.x +
                               (size_t)(mBlock + grow) * DIMN + k0 + gcol;
            asm volatile("global_load_async_to_lds_b128 %0, %1, off"
                         :: "v"(ldsA[buf]), "v"(xp) : "memory");
            asm volatile("global_load_async_to_lds_b128 %0, %1, off offset:16"
                         :: "v"(ldsA[buf]), "v"(xp) : "memory");
        } else {
            const float* xp = (const float*)jb.x +
                              (size_t)(mBlock + grow) * DIMN + k0 + gcol;
            v4f f0 = *(const v4f*)(xp + 0);
            v4f f1 = *(const v4f*)(xp + 4);
            v4f f2 = *(const v4f*)(xp + 8);
            v4f f3 = *(const v4f*)(xp + 12);
            v8h h0, h1;
            for (int e = 0; e < 4; ++e) {
                h0[e]     = (half_t)f0[e];
                h0[e + 4] = (half_t)f1[e];
                h1[e]     = (half_t)f2[e];
                h1[e + 4] = (half_t)f3[e];
            }
            *(v8h*)&As[buf][grow * 40 + gcol]     = h0;
            *(v8h*)&As[buf][grow * 40 + gcol + 8] = h1;
            if (k0 + 32 < DIMN) __builtin_prefetch(xp + 32, 0, 0);
        }
        // ---- W tile (always f32 -> f16) ----
        {
            const float* wp = jb.W + (size_t)(nBlock + grow) * DIMN + k0 + gcol;
            v4f f0 = *(const v4f*)(wp + 0);
            v4f f1 = *(const v4f*)(wp + 4);
            v4f f2 = *(const v4f*)(wp + 8);
            v4f f3 = *(const v4f*)(wp + 12);
            v8h h0, h1;
            for (int e = 0; e < 4; ++e) {
                h0[e]     = (half_t)f0[e];
                h0[e + 4] = (half_t)f1[e];
                h1[e]     = (half_t)f2[e];
                h1[e + 4] = (half_t)f3[e];
            }
            *(v8h*)&Bs[buf][grow * 40 + gcol]     = h0;
            *(v8h*)&Bs[buf][grow * 40 + gcol + 8] = h1;
            if (k0 + 32 < DIMN) __builtin_prefetch(wp + 32, 0, 0);
        }
    };

    v8f acc[2][4];
    const v8f vzero = {0.f,0.f,0.f,0.f,0.f,0.f,0.f,0.f};
    for (int i = 0; i < 2; ++i)
        for (int j = 0; j < 4; ++j) acc[i][j] = vzero;

    stage(0, 0);
    for (int k0 = 0; k0 < DIMN; k0 += 32) {
        const int buf = (k0 >> 5) & 1;
        if (XHALF) asm volatile("s_wait_asynccnt 0" ::: "memory");
        __syncthreads();
        if (k0 + 32 < DIMN) stage(buf ^ 1, k0 + 32);   // overlaps with WMMAs below

        // ---- A fragments: 16x32 f16, two contiguous 8-half chunks per lane
        Frag16 af[2];
        for (int i = 0; i < 2; ++i) {
            const int row = wm + i * 16 + ll;
            af[i].h[0] = *(const v8h*)&As[buf][row * 40 + lh * 8];
            af[i].h[1] = *(const v8h*)&As[buf][row * 40 + lh * 8 + 16];
        }
        // ---- B fragments: 32x16 f16, two contiguous 8-half chunks per lane
        Frag16 bf[4];
        for (int j = 0; j < 4; ++j) {
            const int nr = wn + j * 16 + ll;
            bf[j].h[0] = *(const v8h*)&Bs[buf][nr * 40 + lh * 8];
            bf[j].h[1] = *(const v8h*)&Bs[buf][nr * 40 + lh * 8 + 16];
        }
        for (int i = 0; i < 2; ++i)
            for (int j = 0; j < 4; ++j)
                acc[i][j] = wmma_f16(af[i].v, bf[j].v, acc[i][j]);
    }

    // ---- epilogue: bias + scale + layout-specific store ----
    for (int j = 0; j < 4; ++j) {
        const int n = nBlock + wn + j * 16 + ll;
        const float bv = jb.bias[n];
        for (int i = 0; i < 2; ++i) {
            for (int v = 0; v < 8; ++v) {
                const int m = mBlock + wm + i * 16 + v + lh * 8;
                const float val = (acc[i][j][v] + bv) * jb.scale;
                if (MODE == 2) {
                    ((float*)jb.out)[(size_t)m * DIMN + n] = val;
                } else {
                    const int bb = m >> 11, l = m & 2047;
                    const int hh = n >> 6,  d = n & 63;
                    if (MODE == 0)
                        ((half_t*)jb.out)[(((size_t)(bb * HEADS + hh) * SEQ + l) << 6) + d]
                            = (half_t)val;
                    else
                        ((half_t*)jb.out)[(((size_t)(bb * HEADS + hh) * HDIM + d) * SEQ) + l]
                            = (half_t)val;
                }
            }
        }
    }
}

// ---------------------------------------------------------------------------
// Flash attention: S = Qr*Kr^T + Qi*Ki^T (Q pre-scaled), online softmax,
// Or = P*Vr, Oi = P*Vi.  One wave owns 16 q-rows; KV loop in steps of 32.
// All 32 B-fragment loads of a slab are issued before any WMMA so the
// waits stagger and V-loads complete under the S-WMMAs + softmax VALU work.
// ---------------------------------------------------------------------------
__global__ __launch_bounds__(256)
void attn_kernel(const half_t* __restrict__ Qr, const half_t* __restrict__ Qi,
                 const half_t* __restrict__ Kr, const half_t* __restrict__ Ki,
                 const half_t* __restrict__ VrT, const half_t* __restrict__ ViT,
                 half_t* __restrict__ Or, half_t* __restrict__ Oi) {
    __shared__ half_t pb[8][16 * 40];   // per-wave P staging (C-layout -> A-layout)

    const int tid  = threadIdx.x;
    const int lane = tid & 31;
    const int wave = tid >> 5;
    const int ll   = lane & 15;
    const int lh   = lane >> 4;
    const int bh   = blockIdx.y;            // b*16 + h
    const int b    = bh >> 4;
    const int h    = bh & 15;
    const int q0   = blockIdx.x * 128 + wave * 16;

    const size_t headOff  = (size_t)bh * SEQ * HDIM;  // Q,K base
    const size_t headOffT = (size_t)bh * HDIM * SEQ;  // V^T base
    half_t* pw = pb[wave];

    // Q fragments (held in registers for whole KV loop): 2 d-chunks x {r,i}
    Frag16 qr[2], qi[2];
    for (int c = 0; c < 2; ++c) {
        const half_t* qpr = Qr + headOff + (size_t)(q0 + ll) * HDIM + c * 32 + lh * 8;
        const half_t* qpi = Qi + headOff + (size_t)(q0 + ll) * HDIM + c * 32 + lh * 8;
        qr[c].h[0] = *(const v8h*)qpr;
        qr[c].h[1] = *(const v8h*)(qpr + 16);
        qi[c].h[0] = *(const v8h*)qpi;
        qi[c].h[1] = *(const v8h*)(qpi + 16);
    }

    const v8f vzero = {0.f,0.f,0.f,0.f,0.f,0.f,0.f,0.f};
    v8f o_r[4], o_i[4];
    for (int j = 0; j < 4; ++j) { o_r[j] = vzero; o_i[j] = vzero; }
    float mrow[8], lrow[8];
    for (int v = 0; v < 8; ++v) { mrow[v] = -1e30f; lrow[v] = 0.f; }

    for (int kv = 0; kv < SEQ; kv += 32) {
        // ---- issue ALL slab loads up front (32x global_load_b128 in flight) ----
        Frag16 bkr[2][2], bki[2][2];
        for (int j = 0; j < 2; ++j) {          // kv half (16 cols)
            for (int c = 0; c < 2; ++c) {      // d contraction chunk (32)
                const size_t ko = headOff + (size_t)(kv + j * 16 + ll) * HDIM
                                + c * 32 + lh * 8;
                bkr[j][c].h[0] = *(const v8h*)(Kr + ko);
                bkr[j][c].h[1] = *(const v8h*)(Kr + ko + 16);
                bki[j][c].h[0] = *(const v8h*)(Ki + ko);
                bki[j][c].h[1] = *(const v8h*)(Ki + ko + 16);
            }
        }
        Frag16 bvr[4], bvi[4];
        for (int j = 0; j < 4; ++j) {
            const size_t vo = headOffT + (size_t)(j * 16 + ll) * SEQ + kv + lh * 8;
            bvr[j].h[0] = *(const v8h*)(VrT + vo);
            bvr[j].h[1] = *(const v8h*)(VrT + vo + 16);
            bvi[j].h[0] = *(const v8h*)(ViT + vo);
            bvi[j].h[1] = *(const v8h*)(ViT + vo + 16);
        }

        // ---- S = Qr*Kr^T + Qi*Ki^T over this 32-wide KV slab ----
        v8f s[2];
        s[0] = vzero; s[1] = vzero;
        for (int j = 0; j < 2; ++j)
            for (int c = 0; c < 2; ++c) {
                s[j] = wmma_f16(qr[c].v, bkr[j][c].v, s[j]);
                s[j] = wmma_f16(qi[c].v, bki[j][c].v, s[j]);
            }

        // ---- online softmax (rows live in 16-lane halves) ----
        for (int v = 0; v < 8; ++v) {
            float mx = fmaxf(s[0][v], s[1][v]);
            mx = fmaxf(mx, __shfl_xor(mx, 1, 32));
            mx = fmaxf(mx, __shfl_xor(mx, 2, 32));
            mx = fmaxf(mx, __shfl_xor(mx, 4, 32));
            mx = fmaxf(mx, __shfl_xor(mx, 8, 32));
            const float mnew  = fmaxf(mrow[v], mx);
            const float alpha = __expf(mrow[v] - mnew);
            mrow[v] = mnew;
            const float p0 = __expf(s[0][v] - mnew);
            const float p1 = __expf(s[1][v] - mnew);
            s[0][v] = p0; s[1][v] = p1;
            float rs = p0 + p1;
            rs += __shfl_xor(rs, 1, 32);
            rs += __shfl_xor(rs, 2, 32);
            rs += __shfl_xor(rs, 4, 32);
            rs += __shfl_xor(rs, 8, 32);
            lrow[v] = lrow[v] * alpha + rs;
            for (int j = 0; j < 4; ++j) { o_r[j][v] *= alpha; o_i[j][v] *= alpha; }
        }

        // ---- P: C-layout (f32) -> LDS (f16) -> A-fragment ----
        for (int v = 0; v < 8; ++v) {
            const int row = v + lh * 8;
            pw[row * 40 + ll]      = (half_t)s[0][v];
            pw[row * 40 + 16 + ll] = (half_t)s[1][v];
        }
        asm volatile("s_wait_dscnt 0" ::: "memory");
        Frag16 pa;
        pa.h[0] = *(const v8h*)&pw[ll * 40 + lh * 8];
        pa.h[1] = *(const v8h*)&pw[ll * 40 + lh * 8 + 16];

        // ---- O += P * V  (V fragments already resident) ----
        for (int j = 0; j < 4; ++j) {
            o_r[j] = wmma_f16(pa.v, bvr[j].v, o_r[j]);
            o_i[j] = wmma_f16(pa.v, bvi[j].v, o_i[j]);
        }
    }

    // ---- normalize + store f16 [B,L,DIM] ----
    for (int v = 0; v < 8; ++v) lrow[v] = 1.f / lrow[v];
    for (int j = 0; j < 4; ++j) {
        for (int v = 0; v < 8; ++v) {
            const int q = q0 + v + lh * 8;
            const int d = j * 16 + ll;
            const size_t oidx = ((size_t)(b * SEQ + q)) * DIMN + h * HDIM + d;
            Or[oidx] = (half_t)(o_r[j][v] * lrow[v]);
            Oi[oidx] = (half_t)(o_i[j][v] * lrow[v]);
        }
    }
}

// ---------------------------------------------------------------------------
extern "C" void kernel_launch(void* const* d_in, const int* in_sizes, int n_in,
                              void* d_out, int out_size, void* d_ws, size_t ws_size,
                              hipStream_t stream) {
    (void)in_sizes; (void)n_in; (void)out_size; (void)ws_size;

    // d_in order: q_r q_i k_r k_i v_r v_i,
    //   Wq_r bq_r Wq_i bq_i Wk_r bk_r Wk_i bk_i Wv_r bv_r Wv_i bv_i Wo_r bo_r Wo_i bo_i
    const float* xs[6] = {(const float*)d_in[0], (const float*)d_in[1],
                          (const float*)d_in[2], (const float*)d_in[3],
                          (const float*)d_in[4], (const float*)d_in[5]};
    const float* Ws[6] = {(const float*)d_in[6],  (const float*)d_in[8],
                          (const float*)d_in[10], (const float*)d_in[12],
                          (const float*)d_in[14], (const float*)d_in[16]};
    const float* bs[6] = {(const float*)d_in[7],  (const float*)d_in[9],
                          (const float*)d_in[11], (const float*)d_in[13],
                          (const float*)d_in[15], (const float*)d_in[17]};
    const float* Wo_r = (const float*)d_in[18];
    const float* bo_r = (const float*)d_in[19];
    const float* Wo_i = (const float*)d_in[20];
    const float* bo_i = (const float*)d_in[21];

    // workspace: 8 f16 buffers of 8,388,608 elements each (~134 MB total)
    const size_t S = (size_t)BATCH * HEADS * SEQ * HDIM;   // == B*L*DIM
    half_t* ws  = (half_t*)d_ws;
    half_t* Qr  = ws + 0 * S;
    half_t* Qi  = ws + 1 * S;
    half_t* Kr  = ws + 2 * S;
    half_t* Ki  = ws + 3 * S;
    half_t* VrT = ws + 4 * S;
    half_t* ViT = ws + 5 * S;
    half_t* Orh = ws + 6 * S;
    half_t* Oih = ws + 7 * S;

    // ---- Q/K projections (mode 0, Q pre-scaled by HD^-0.5) ----
    GemmArgs gqk;
    half_t* qkouts[4] = {Qr, Qi, Kr, Ki};
    const float qkscale[4] = {SCALEQ, SCALEQ, 1.f, 1.f};
    for (int j = 0; j < 4; ++j) {
        gqk.job[j].x = xs[j];    gqk.job[j].W = Ws[j];
        gqk.job[j].bias = bs[j]; gqk.job[j].out = qkouts[j];
        gqk.job[j].scale = qkscale[j];
    }
    for (int j = 4; j < 6; ++j) gqk.job[j] = gqk.job[0];
    proj_gemm<0, false><<<dim3(DIMN / 128, MTOT / 128, 4), 256, 0, stream>>>(gqk);

    // ---- V projections (mode 1: transposed f16 output) ----
    GemmArgs gv;
    half_t* vouts[2] = {VrT, ViT};
    for (int j = 0; j < 2; ++j) {
        gv.job[j].x = xs[4 + j];    gv.job[j].W = Ws[4 + j];
        gv.job[j].bias = bs[4 + j]; gv.job[j].out = vouts[j];
        gv.job[j].scale = 1.f;
    }
    for (int j = 2; j < 6; ++j) gv.job[j] = gv.job[0];
    proj_gemm<1, false><<<dim3(DIMN / 128, MTOT / 128, 2), 256, 0, stream>>>(gv);

    // ---- attention ----
    attn_kernel<<<dim3(SEQ / 128, BATCH * HEADS), 256, 0, stream>>>(
        Qr, Qi, Kr, Ki, VrT, ViT, Orh, Oih);

    // ---- output projections (mode 2, f16 A staged via async-to-LDS) ----
    GemmArgs go;
    go.job[0].x = Orh; go.job[0].W = Wo_r; go.job[0].bias = bo_r;
    go.job[0].out = (float*)d_out;                       go.job[0].scale = 1.f;
    go.job[1].x = Oih; go.job[1].W = Wo_i; go.job[1].bias = bo_i;
    go.job[1].out = (float*)d_out + (size_t)MTOT * DIMN; go.job[1].scale = 1.f;
    for (int j = 2; j < 6; ++j) go.job[j] = go.job[0];
    proj_gemm<2, true><<<dim3(DIMN / 128, MTOT / 128, 2), 256, 0, stream>>>(go);
}
